// ATSAttention_70600672412334
// MI455X (gfx1250) — matrix operand order repaired and checked by reading, not compiled, not run on hardware
//
#include <hip/hip_runtime.h>

// ---------------- problem constants ----------------
#define DIMC   768
#define NHEAD  12
#define HD     64
#define BATCH  32
#define NTOK   577
#define NPAD   608          // 577 padded to multiple of 32
#define SCNT   256
#define SP1    257
#define MQKV   (BATCH*NTOK) // 18464 = 16*1154
#define SCALE_Q 0.125f      // 64^-0.5
#define NEGBIG (-3.402823466e38f)

// ---------------- WMMA fragment types ----------------
typedef __attribute__((ext_vector_type(16))) __bf16        v16bf;
typedef __attribute__((ext_vector_type(8)))  float         v8f;
typedef __attribute__((ext_vector_type(4)))  unsigned int  uiv4;

union FragBF { v16bf v; uiv4 q[2]; };

static __device__ __forceinline__ v8f wmma_bf16(const FragBF& a, const FragBF& b, v8f c) {
  return __builtin_amdgcn_wmma_f32_16x16x32_bf16(false, a.v, false, b.v, (short)0, c, false, false);
}

static __device__ __forceinline__ unsigned short f2bf(float f) {
  unsigned int u = __float_as_uint(f);
  u += 0x7FFFu + ((u >> 16) & 1u);          // round-to-nearest-even
  return (unsigned short)(u >> 16);
}
static __device__ __forceinline__ float bf2f(unsigned short h) {
  return __uint_as_float(((unsigned int)h) << 16);
}

// ---------------- elementwise helpers ----------------
__global__ void cvt_f32_bf16_kernel(const float* __restrict__ src,
                                    unsigned short* __restrict__ dst, int n) {
  int i = blockIdx.x * 256 + threadIdx.x;
  if (i < n) dst[i] = f2bf(src[i]);
}

__global__ void zero_f32_kernel(float* __restrict__ p, int n) {
  int i = blockIdx.x * 256 + threadIdx.x;
  if (i < n) p[i] = 0.f;
}

// ---------------- kernel 1: QKV GEMM  (x[M,768] * qkv_w[d,768]^T) ----------------
// grid (1154, 9), block 128 (4 waves). Each wave computes a 16x64 C block
// (4 accumulators sharing one A fragment). K-loop is double-buffered so the
// next chunk's 10x16B loads are in flight behind 4 back-to-back WMMAs.
__global__ __launch_bounds__(128) void qkv_gemm_kernel(
    const unsigned short* __restrict__ xbf, const unsigned short* __restrict__ wbf,
    unsigned short* __restrict__ qws, unsigned short* __restrict__ kws,
    unsigned short* __restrict__ vtws) {
  const int lane = threadIdx.x & 31;
  const int wv   = threadIdx.x >> 5;
  const int half = lane >> 4;
  const int l15  = lane & 15;
  const int mt   = blockIdx.x;
  const int nt64 = blockIdx.y * 4 + wv;     // 0..35 (tile of 64 output cols)
  const int m = mt * 16 + l15;
  const unsigned short* arow = xbf + (size_t)m * DIMC;
  const unsigned short* brow0 = wbf + (size_t)(nt64 * 64 +  0 + l15) * DIMC;
  const unsigned short* brow1 = wbf + (size_t)(nt64 * 64 + 16 + l15) * DIMC;
  const unsigned short* brow2 = wbf + (size_t)(nt64 * 64 + 32 + l15) * DIMC;
  const unsigned short* brow3 = wbf + (size_t)(nt64 * 64 + 48 + l15) * DIMC;
  const int ka = half * 16;                 // A-matrix lane K pattern
  const int kb = half * 16;                 // B-matrix lane K pattern
  v8f acc0 = {}, acc1 = {}, acc2 = {}, acc3 = {};
  FragBF a[2], f0[2], f1[2], f2[2], f3[2];
  auto ldc = [&](int c, int s) {
    a[s].q[0]  = *(const uiv4*)(arow  + c * 32 + ka);
    a[s].q[1]  = *(const uiv4*)(arow  + c * 32 + ka + 8);
    f0[s].q[0] = *(const uiv4*)(brow0 + c * 32 + kb);
    f0[s].q[1] = *(const uiv4*)(brow0 + c * 32 + kb + 8);
    f1[s].q[0] = *(const uiv4*)(brow1 + c * 32 + kb);
    f1[s].q[1] = *(const uiv4*)(brow1 + c * 32 + kb + 8);
    f2[s].q[0] = *(const uiv4*)(brow2 + c * 32 + kb);
    f2[s].q[1] = *(const uiv4*)(brow2 + c * 32 + kb + 8);
    f3[s].q[0] = *(const uiv4*)(brow3 + c * 32 + kb);
    f3[s].q[1] = *(const uiv4*)(brow3 + c * 32 + kb + 8);
  };
  ldc(0, 0);
  #pragma unroll
  for (int c = 0; c < DIMC / 32; ++c) {
    const int cur = c & 1;
    if (c + 1 < DIMC / 32) ldc(c + 1, cur ^ 1);
    acc0 = wmma_bf16(a[cur], f0[cur], acc0);
    acc1 = wmma_bf16(a[cur], f1[cur], acc1);
    acc2 = wmma_bf16(a[cur], f2[cur], acc2);
    acc3 = wmma_bf16(a[cur], f3[cur], acc3);
  }
  // C layout: lane%16 = N col, row = mt*16 + half*8 + rr
  const int mrow0 = mt * 16 + half * 8;
  const v8f* accs[4] = { &acc0, &acc1, &acc2, &acc3 };
  #pragma unroll
  for (int rr = 0; rr < 8; ++rr) {
    const int mm = mrow0 + rr;
    const int bi = mm / NTOK;
    const int n  = mm - bi * NTOK;
    #pragma unroll
    for (int t = 0; t < 4; ++t) {
      const int d = nt64 * 64 + t * 16 + l15;
      const int which = d / DIMC;           // 0=q 1=k 2=v
      const int rem = d - which * DIMC;
      const int h = rem >> 6;
      const int e = rem & 63;
      const size_t bh = (size_t)(bi * NHEAD + h);
      const unsigned short val = f2bf((*accs[t])[rr]);
      if (which == 0)      qws[(bh * NPAD + n) * HD + e] = val;
      else if (which == 1) kws[(bh * NPAD + n) * HD + e] = val;
      else                 vtws[(bh * HD + e) * NPAD + n] = val;
    }
  }
}

// ---------------- kernel 2: cls-row softmax + significance scores ----------------
// grid (12, 32), block 128. sig[b][j-1] += attn[b,h,0,j] * ||v[b,h,j,:]||
__global__ __launch_bounds__(128) void cls_sig_kernel(
    const unsigned short* __restrict__ qws, const unsigned short* __restrict__ kws,
    const unsigned short* __restrict__ vtws, const unsigned char* __restrict__ mask,
    float* __restrict__ sig) {
  const int h = blockIdx.x, b = blockIdx.y;
  const size_t bh = (size_t)(b * NHEAD + h);
  const unsigned short* qb = qws + bh * NPAD * HD;     // token 0 row
  const unsigned short* kb = kws + bh * NPAD * HD;
  const unsigned short* vt = vtws + bh * HD * NPAD;
  __shared__ float q0[HD];
  __shared__ float sc[NTOK];
  __shared__ float red[128];
  const int tid = threadIdx.x;
  if (tid < HD) q0[tid] = bf2f(qb[tid]);
  __syncthreads();
  const bool mi = mask[b * NTOK] != 0;
  float lmax = NEGBIG;
  for (int j = tid; j < NTOK; j += 128) {
    const unsigned short* kr = kb + (size_t)j * HD;
    float dot = 0.f;
    #pragma unroll 8
    for (int e = 0; e < HD; ++e) dot += q0[e] * bf2f(kr[e]);
    float v = (mi && mask[b * NTOK + j]) ? dot * SCALE_Q : NEGBIG;
    sc[j] = v;
    lmax = fmaxf(lmax, v);
  }
  red[tid] = lmax; __syncthreads();
  for (int o = 64; o > 0; o >>= 1) { if (tid < o) red[tid] = fmaxf(red[tid], red[tid + o]); __syncthreads(); }
  const float mx = red[0]; __syncthreads();
  float ls = 0.f;
  for (int j = tid; j < NTOK; j += 128) ls += __expf(sc[j] - mx);
  red[tid] = ls; __syncthreads();
  for (int o = 64; o > 0; o >>= 1) { if (tid < o) red[tid] += red[tid + o]; __syncthreads(); }
  const float inv = 1.f / red[0];
  for (int j = 1 + tid; j < NTOK; j += 128) {
    float p = __expf(sc[j] - mx) * inv;
    float nv = 0.f;
    #pragma unroll 8
    for (int e = 0; e < HD; ++e) { float vv = bf2f(vt[(size_t)e * NPAD + j]); nv += vv * vv; }
    atomicAdd(&sig[b * (NTOK - 1) + (j - 1)], p * sqrtf(nv));
  }
}

// ---------------- kernel 3: CDF sampling + unique-pad ----------------
__device__ __forceinline__ void bitonic_sort256(int* ids, int tid) {
  for (int k = 2; k <= 256; k <<= 1) {
    for (int jj = k >> 1; jj > 0; jj >>= 1) {
      int ixj = tid ^ jj;
      if (ixj > tid) {
        int a = ids[tid], c = ids[ixj];
        bool up = ((tid & k) == 0);
        if (up ? (a > c) : (a < c)) { ids[tid] = c; ids[ixj] = a; }
      }
      __syncthreads();
    }
  }
}

__global__ __launch_bounds__(256) void sample_kernel(
    const float* __restrict__ sig, const unsigned char* __restrict__ mask,
    int* __restrict__ uniq, float* __restrict__ out_mask, float* __restrict__ out_uniq) {
  const int b = blockIdx.x, tid = threadIdx.x;
  __shared__ float cdf[NTOK - 1];
  __shared__ int ids[SCNT];
  if (tid == 0) {
    float acc = 0.f;
    for (int j = 0; j < NTOK - 1; ++j) acc += sig[b * (NTOK - 1) + j];
    const float inv = 1.f / (acc + 1e-6f);
    float c = 0.f;
    for (int j = 0; j < NTOK - 1; ++j) {
      c += sig[b * (NTOK - 1) + j] * inv;
      cdf[j] = mask[b * NTOK + j + 1] ? c : c + 0.1f;
    }
  }
  __syncthreads();
  // inverse-CDF: first-occurrence argmin (matches jnp.argmin)
  const float step = (2.f * tid + 1.f) / (2.f * SCNT);
  float best = 3.402823466e38f; int bj = 0;
  for (int j = 0; j < NTOK - 1; ++j) {
    float d = fabsf(step - cdf[j]);
    if (d < best) { best = d; bj = j; }
  }
  ids[tid] = bj + 1;
  __syncthreads();
  bitonic_sort256(ids, tid);
  int v = ids[tid];
  int pv = (tid > 0) ? ids[tid - 1] : -1;
  __syncthreads();
  if (tid > 0 && v == pv) ids[tid] = NTOK;   // sentinel
  __syncthreads();
  bitonic_sort256(ids, tid);
  int idv = ids[tid];
  if (idv == NTOK) idv = 0;
  uniq[b * SP1 + 1 + tid]     = idv;
  out_uniq[b * SP1 + 1 + tid] = (float)idv;
  out_mask[b * SP1 + 1 + tid] = (idv != 0) ? 1.f : 0.f;
  if (tid == 0) { uniq[b * SP1] = 0; out_uniq[b * SP1] = 0.f; out_mask[b * SP1] = 1.f; }
}

// ---------------- kernel 4: fused gathered attention (flash-style) ----------------
// grid (17, 12, 32) = (row-tile, h, b), block 128 (4 waves).
// Phase 1: scores Qsel·Kᵀ via WMMA → LDS(f32). Softmax per row → LDS P (bf16).
// Phase 2: P·V via WMMA (V stored transposed), write ctx bf16 (b,s,h*64+d).
// Both WMMA phases are double-buffered (prefetch next K/V fragments).
__global__ __launch_bounds__(128) void attn_gather_kernel(
    const unsigned short* __restrict__ qws, const unsigned short* __restrict__ kws,
    const unsigned short* __restrict__ vtws, const int* __restrict__ uniq,
    const unsigned char* __restrict__ mask, unsigned short* __restrict__ ctx) {
  const int tile = blockIdx.x, h = blockIdx.y, b = blockIdx.z;
  const size_t bh = (size_t)(b * NHEAD + h);
  const unsigned short* qb = qws + bh * NPAD * HD;
  const unsigned short* kb = kws + bh * NPAD * HD;
  const unsigned short* vt = vtws + bh * HD * NPAD;
  __shared__ __align__(16) float          Sbuf[16][592];
  __shared__ __align__(16) unsigned short Pbuf[16][NPAD];
  const int lane = threadIdx.x & 31;
  const int wv   = threadIdx.x >> 5;
  const int half = lane >> 4;
  const int l15  = lane & 15;
  const int ka = half * 16;
  const int kbk = half * 16;

  // ---- phase 1: 16 gathered q-rows vs all k, j-tiles split over 4 waves ----
  {
    int s = tile * 16 + l15;
    int tok = (s < SP1) ? uniq[b * SP1 + s] : 0;
    const unsigned short* qrow = qb + (size_t)tok * HD;
    FragBF a0, a1;
    a0.q[0] = *(const uiv4*)(qrow + ka);
    a0.q[1] = *(const uiv4*)(qrow + ka + 8);
    a1.q[0] = *(const uiv4*)(qrow + 32 + ka);
    a1.q[1] = *(const uiv4*)(qrow + 32 + ka + 8);
    FragBF kf0[2], kf1[2];
    auto ldk = [&](int jt, int s2) {
      const unsigned short* krow = kb + (size_t)(jt * 16 + l15) * HD;
      kf0[s2].q[0] = *(const uiv4*)(krow + kbk);
      kf0[s2].q[1] = *(const uiv4*)(krow + kbk + 8);
      kf1[s2].q[0] = *(const uiv4*)(krow + 32 + kbk);
      kf1[s2].q[1] = *(const uiv4*)(krow + 32 + kbk + 8);
    };
    ldk(wv, 0);
    #pragma unroll
    for (int it = 0; it < 10; ++it) {
      const int jt = wv + it * 4;
      if (jt >= 37) break;
      const int cur = it & 1;
      int nj = jt + 4; if (nj >= 37) nj = jt;      // clamped prefetch (valid addr)
      ldk(nj, cur ^ 1);
      v8f c = {};
      c = wmma_bf16(a0, kf0[cur], c);
      c = wmma_bf16(a1, kf1[cur], c);
      const int jbase = jt * 16;
      const int m0 = half * 8;
      #pragma unroll
      for (int rr = 0; rr < 8; ++rr)
        Sbuf[m0 + rr][jbase + l15] = c[rr] * SCALE_Q;
    }
  }
  __syncthreads();

  // ---- softmax: wave wv handles rows 4wv..4wv+3 ----
  for (int row = wv * 4; row < wv * 4 + 4; ++row) {
    int s = tile * 16 + row;
    int tok = (s < SP1) ? uniq[b * SP1 + s] : 0;
    const bool mi = mask[b * NTOK + tok] != 0;
    float mx = NEGBIG;
    for (int j = lane; j < NTOK; j += 32) {
      float v = (mi && mask[b * NTOK + j]) ? Sbuf[row][j] : NEGBIG;
      Sbuf[row][j] = v;
      mx = fmaxf(mx, v);
    }
    #pragma unroll
    for (int o = 16; o > 0; o >>= 1) mx = fmaxf(mx, __shfl_xor(mx, o, 32));
    float ls = 0.f;
    for (int j = lane; j < NTOK; j += 32) ls += __expf(Sbuf[row][j] - mx);
    #pragma unroll
    for (int o = 16; o > 0; o >>= 1) ls += __shfl_xor(ls, o, 32);
    const float inv = 1.f / ls;
    for (int j = lane; j < NPAD; j += 32) {
      float p = (j < NTOK) ? __expf(Sbuf[row][j] - mx) * inv : 0.f;
      Pbuf[row][j] = f2bf(p);
    }
  }
  __syncthreads();

  // ---- phase 2: out(16x64) = P(16x608) * V(608x64); wave wv -> d-tile wv ----
  {
    const int dcol = wv * 16 + l15;
    const unsigned short* vrow = vt + (size_t)dcol * NPAD;   // V^T row d, contiguous in j
    FragBF pf[2], vf[2];
    auto ldpv = [&](int c, int s2) {
      const int jb = c * 32;
      pf[s2].q[0] = *(const uiv4*)(&Pbuf[l15][jb + ka]);
      pf[s2].q[1] = *(const uiv4*)(&Pbuf[l15][jb + ka + 8]);
      vf[s2].q[0] = *(const uiv4*)(vrow + jb + kbk);
      vf[s2].q[1] = *(const uiv4*)(vrow + jb + kbk + 8);
    };
    ldpv(0, 0);
    v8f acc = {};
    #pragma unroll
    for (int c = 0; c < NPAD / 32; ++c) {
      const int cur = c & 1;
      if (c + 1 < NPAD / 32) ldpv(c + 1, cur ^ 1);
      acc = wmma_bf16(pf[cur], vf[cur], acc);
    }
    const int m0 = half * 8;
    #pragma unroll
    for (int rr = 0; rr < 8; ++rr) {
      int srow = tile * 16 + m0 + rr;
      if (srow < SP1)
        ctx[((size_t)b * SP1 + srow) * DIMC + h * HD + dcol] = f2bf(acc[rr]);
    }
  }
}

// ---------------- kernel 5: output projection GEMM + bias ----------------
// grid (514, 3), block 128. Each wave computes 16x64 of out = ctx·proj_wᵀ + b.
__global__ __launch_bounds__(128) void proj_gemm_kernel(
    const unsigned short* __restrict__ ctx, const unsigned short* __restrict__ wbf,
    const float* __restrict__ bias, float* __restrict__ out) {
  const int lane = threadIdx.x & 31;
  const int wv   = threadIdx.x >> 5;
  const int half = lane >> 4;
  const int l15  = lane & 15;
  const int mt   = blockIdx.x;
  const int nt64 = blockIdx.y * 4 + wv;          // 0..11
  const unsigned short* arow = ctx + (size_t)(mt * 16 + l15) * DIMC;
  const unsigned short* brow0 = wbf + (size_t)(nt64 * 64 +  0 + l15) * DIMC;
  const unsigned short* brow1 = wbf + (size_t)(nt64 * 64 + 16 + l15) * DIMC;
  const unsigned short* brow2 = wbf + (size_t)(nt64 * 64 + 32 + l15) * DIMC;
  const unsigned short* brow3 = wbf + (size_t)(nt64 * 64 + 48 + l15) * DIMC;
  const int ka = half * 16;
  const int kb = half * 16;
  v8f acc0 = {}, acc1 = {}, acc2 = {}, acc3 = {};
  FragBF a[2], f0[2], f1[2], f2[2], f3[2];
  auto ldc = [&](int c, int s) {
    a[s].q[0]  = *(const uiv4*)(arow  + c * 32 + ka);
    a[s].q[1]  = *(const uiv4*)(arow  + c * 32 + ka + 8);
    f0[s].q[0] = *(const uiv4*)(brow0 + c * 32 + kb);
    f0[s].q[1] = *(const uiv4*)(brow0 + c * 32 + kb + 8);
    f1[s].q[0] = *(const uiv4*)(brow1 + c * 32 + kb);
    f1[s].q[1] = *(const uiv4*)(brow1 + c * 32 + kb + 8);
    f2[s].q[0] = *(const uiv4*)(brow2 + c * 32 + kb);
    f2[s].q[1] = *(const uiv4*)(brow2 + c * 32 + kb + 8);
    f3[s].q[0] = *(const uiv4*)(brow3 + c * 32 + kb);
    f3[s].q[1] = *(const uiv4*)(brow3 + c * 32 + kb + 8);
  };
  ldc(0, 0);
  #pragma unroll
  for (int c = 0; c < DIMC / 32; ++c) {
    const int cur = c & 1;
    if (c + 1 < DIMC / 32) ldc(c + 1, cur ^ 1);
    acc0 = wmma_bf16(a[cur], f0[cur], acc0);
    acc1 = wmma_bf16(a[cur], f1[cur], acc1);
    acc2 = wmma_bf16(a[cur], f2[cur], acc2);
    acc3 = wmma_bf16(a[cur], f3[cur], acc3);
  }
  const int m0 = mt * 16 + half * 8;
  const v8f* accs[4] = { &acc0, &acc1, &acc2, &acc3 };
  #pragma unroll
  for (int t = 0; t < 4; ++t) {
    const int dcol = nt64 * 64 + t * 16 + l15;
    const float bv = bias[dcol];
    #pragma unroll
    for (int rr = 0; rr < 8; ++rr)
      out[(size_t)(m0 + rr) * DIMC + dcol] = (*accs[t])[rr] + bv;
  }
}

// ---------------- host orchestration ----------------
extern "C" void kernel_launch(void* const* d_in, const int* in_sizes, int n_in,
                              void* d_out, int out_size, void* d_ws, size_t ws_size,
                              hipStream_t stream) {
  (void)in_sizes; (void)n_in; (void)out_size; (void)ws_size;
  const float*         x      = (const float*)d_in[0];
  const unsigned char* mask   = (const unsigned char*)d_in[1];
  const float*         qkv_w  = (const float*)d_in[2];
  const float*         proj_w = (const float*)d_in[3];
  const float*         proj_b = (const float*)d_in[4];

  // workspace carve-up (all 256B aligned)
  char* ws = (char*)d_ws;
  size_t off = 0;
  auto carve = [&](size_t bytes) { char* p = ws + off; off += (bytes + 255) & ~(size_t)255; return p; };
  unsigned short* xbf   = (unsigned short*)carve((size_t)MQKV * DIMC * 2);       // 28.4 MB
  unsigned short* wqkv  = (unsigned short*)carve((size_t)3 * DIMC * DIMC * 2);   //  3.5 MB
  unsigned short* wproj = (unsigned short*)carve((size_t)DIMC * DIMC * 2);       //  1.2 MB
  unsigned short* qws   = (unsigned short*)carve((size_t)BATCH * NHEAD * NPAD * HD * 2); // 29.9 MB
  unsigned short* kws   = (unsigned short*)carve((size_t)BATCH * NHEAD * NPAD * HD * 2); // 29.9 MB
  unsigned short* vtws  = (unsigned short*)carve((size_t)BATCH * NHEAD * HD * NPAD * 2); // 29.9 MB
  unsigned short* ctx   = (unsigned short*)carve((size_t)BATCH * SP1 * DIMC * 2);        // 12.6 MB
  float*          sig   = (float*)carve((size_t)BATCH * (NTOK - 1) * 4);
  int*            uniq  = (int*)carve((size_t)BATCH * SP1 * 4);

  // output regions: out (B*257*768) | new_mask (B*257) | uniq (B*257), all as f32
  float* out_f    = (float*)d_out;
  float* out_mask = out_f + (size_t)BATCH * SP1 * DIMC;
  float* out_uniq = out_mask + (size_t)BATCH * SP1;

  // 0) converts + zero
  {
    int n1 = MQKV * DIMC;
    cvt_f32_bf16_kernel<<<(n1 + 255) / 256, 256, 0, stream>>>(x, xbf, n1);
    int n2 = 3 * DIMC * DIMC;
    cvt_f32_bf16_kernel<<<(n2 + 255) / 256, 256, 0, stream>>>(qkv_w, wqkv, n2);
    int n3 = DIMC * DIMC;
    cvt_f32_bf16_kernel<<<(n3 + 255) / 256, 256, 0, stream>>>(proj_w, wproj, n3);
    int n4 = BATCH * (NTOK - 1);
    zero_f32_kernel<<<(n4 + 255) / 256, 256, 0, stream>>>(sig, n4);
  }
  // 1) QKV projection GEMM (WMMA bf16, 16x64 per wave, double-buffered)
  qkv_gemm_kernel<<<dim3(MQKV / 16, 9), 128, 0, stream>>>(xbf, wqkv, qws, kws, vtws);
  // 2) cls-row attention + significance
  cls_sig_kernel<<<dim3(NHEAD, BATCH), 128, 0, stream>>>(qws, kws, vtws, mask, sig);
  // 3) sampling + unique-pad
  sample_kernel<<<BATCH, 256, 0, stream>>>(sig, mask, uniq, out_mask, out_uniq);
  // 4) fused gathered attention (WMMA bf16, double-buffered)
  attn_gather_kernel<<<dim3(17, NHEAD, BATCH), 128, 0, stream>>>(qws, kws, vtws, uniq, mask, ctx);
  // 5) output projection GEMM + bias (WMMA bf16, 16x64 per wave, double-buffered)
  proj_gemm_kernel<<<dim3((BATCH * SP1) / 16, 3), 128, 0, stream>>>(ctx, wproj, proj_b, out_f);
}
